// PointMambaLayer_12850542150376
// MI455X (gfx1250) — matrix operand order repaired and checked by reading, not compiled, not run on hardware
//
#include <hip/hip_runtime.h>
#include <math.h>

// ---------------------------------------------------------------------------
// PointMamba (N=4096, DM=256, DI=DS=512, NH=16, HD=32, L=2) for gfx1250.
// Chunked Mamba2 scan (Q=32, 128 chunks); all GEMM work on v_wmma_f32_16x16x32_f16.
// Projection GEMMs are 2x4 register-blocked (8 WMMA per K-step per wave).
// Sequential-scan B tiles staged to LDS via the Tensor Data Mover.
// ---------------------------------------------------------------------------

#define USE_TDM 1

#define NPTS   4096
#define DMC    256
#define DIC    512
#define DSC    512
#define NHC    16
#define HDC    32
#define DPROJ  2064   // [z(512), xBC(1536), dt(16)]
#define CONVC  1536
#define QC     32     // chunk length
#define NCHUNK 128    // 4096/32
#define SSPLIT 8      // state splits per head
#define SLEN   64     // 512/8 state rows per block
#define EPSF   1e-5f

typedef __attribute__((ext_vector_type(16))) _Float16 v16h;
typedef __attribute__((ext_vector_type(8)))  _Float16 v8h;
typedef __attribute__((ext_vector_type(8)))  float    v8f;
typedef __attribute__((ext_vector_type(4)))  unsigned int v4u;
typedef __attribute__((ext_vector_type(8)))  unsigned int v8u;

// ---------------- WMMA fragment helpers (contiguous-K A/B order, wave32) --------

// A fragment: rows m0..m0+15, K-slice k0..k0+31 of row-major f16 [., ld].
// lane l holds row M=l&15; element e -> K = ((l&16)?16:0) + e  (16 contiguous).
__device__ __forceinline__ v16h frag_a(const _Float16* A, int m0, int k0, int ld, int lane) {
  int m   = m0 + (lane & 15);
  int klo = (lane & 16) ? 16 : 0;
  return *(const v16h*)(A + (size_t)m * ld + k0 + klo);
}

// B fragment for D = A @ W^T: Bfrag[k][n] = W[n][k], W row-major [., ld].
// lane l holds col N=l&15; element e -> K = ((l&16)?16:0) + e  (16 contiguous).
__device__ __forceinline__ v16h frag_w(const _Float16* W, int n0, int k0, int ld, int lane) {
  int n   = n0 + (lane & 15);
  int klo = (lane & 16) ? 16 : 0;
  return *(const v16h*)(W + (size_t)n * ld + k0 + klo);
}

__device__ __forceinline__ v8f wmma_f16(v16h a, v16h b, v8f c) {
  return __builtin_amdgcn_wmma_f32_16x16x32_f16(false, a, false, b, (short)0, c, false, false);
}

// One 16x16 output tile of C[M,N] = A[M,K] @ W[N,K]^T, f32 store.
__device__ __forceinline__ void wave_tile_gemm(const _Float16* __restrict__ A,
                                               const _Float16* __restrict__ W,
                                               float* __restrict__ C,
                                               int m0, int n0, int K,
                                               int ldA, int ldW, int ldC, int lane) {
  v8f acc = {};
  for (int k0 = 0; k0 < K; k0 += 32) {
    v16h af = frag_a(A, m0, k0, ldA, lane);
    v16h bf = frag_w(W, n0, k0, ldW, lane);
    acc = wmma_f16(af, bf, acc);
  }
  int n  = n0 + (lane & 15);
  int mo = (lane & 16) ? 8 : 0;
#pragma unroll
  for (int r = 0; r < 8; ++r)
    C[(size_t)(m0 + r + mo) * ldC + n] = acc[r];
}

__device__ __forceinline__ float siluf(float x) { return x / (1.f + __expf(-x)); }

// ---------------- small elementwise / setup kernels -------------------------

__global__ void __launch_bounds__(256) rank_kernel(const float* __restrict__ p_gs,
                                                   int* __restrict__ rank) {
  __shared__ float zk[NPTS];
  for (int j = threadIdx.x; j < NPTS; j += 256) zk[j] = p_gs[j * 3 + 2];
  __syncthreads();
  int i = blockIdx.x * 256 + threadIdx.x;
  float zi = zk[i];
  int r = 0;
  for (int j = 0; j < NPTS; ++j) {
    float zj = zk[j];
    r += (zj > zi) || (zj == zi && j < i);   // stable argsort of -z (ascending)
  }
  rank[i] = r;
}

__global__ void __launch_bounds__(256) scatter_kernel(const float* __restrict__ f,
                                                      const int* __restrict__ rank,
                                                      float* __restrict__ hid) {
  int idx = blockIdx.x * 256 + threadIdx.x;   // NPTS*DMC
  int i = idx >> 8, d = idx & 255;
  hid[(size_t)rank[i] * DMC + d] = f[idx];
}

__global__ void __launch_bounds__(256) cvt_kernel(const float* __restrict__ src,
                                                  _Float16* __restrict__ dst, int count) {
  int i = blockIdx.x * 256 + threadIdx.x;
  if (i < count) dst[i] = (_Float16)src[i];
}

// residual update + rmsnorm -> f16 activations
__global__ void __launch_bounds__(256) prenorm_kernel(const float* __restrict__ hid,
                                                      float* __restrict__ resid,
                                                      const float* __restrict__ w,
                                                      _Float16* __restrict__ xn16,
                                                      int addFlag) {
  __shared__ float red[256];
  int n = blockIdx.x, d = threadIdx.x;
  float r = hid[(size_t)n * DMC + d] + (addFlag ? resid[(size_t)n * DMC + d] : 0.f);
  resid[(size_t)n * DMC + d] = r;
  red[d] = r * r;
  __syncthreads();
  for (int s = 128; s > 0; s >>= 1) { if (d < s) red[d] += red[d + s]; __syncthreads(); }
  float inv = rsqrtf(red[0] / (float)DMC + EPSF);
  xn16[(size_t)n * DMC + d] = (_Float16)(r * inv * w[d]);
}

// generic wave-per-tile GEMM (used for ragged remainder + G matrices)
__global__ void __launch_bounds__(256) gemm_kernel(const _Float16* __restrict__ A,
                                                   const _Float16* __restrict__ W,
                                                   float* __restrict__ C,
                                                   int Mtiles, int Ntiles, int K,
                                                   int ldA, int ldW, int ldC, int nbase) {
  int wid = blockIdx.x * 8 + (threadIdx.x >> 5);
  if (wid >= Mtiles * Ntiles) return;         // wave-uniform
  int mt = wid / Ntiles, nt = wid % Ntiles;
  wave_tile_gemm(A, W, C, mt * 16, nbase + nt * 16, K, ldA, ldW, ldC, threadIdx.x & 31);
}

// 2x4 register-blocked GEMM: each wave -> 32x64 output tile, 8 WMMA per K-step.
__global__ void __launch_bounds__(256) gemm_blocked_kernel(const _Float16* __restrict__ A,
                                                           const _Float16* __restrict__ W,
                                                           float* __restrict__ Cmat,
                                                           int MB, int NB, int K,
                                                           int ldA, int ldW, int ldC) {
  int wid = blockIdx.x * 8 + (threadIdx.x >> 5);
  if (wid >= MB * NB) return;                 // wave-uniform
  int lane = threadIdx.x & 31;
  int m0 = (wid / NB) * 32;
  int n0 = (wid % NB) * 64;

  v8f acc[2][4];
#pragma unroll
  for (int mi = 0; mi < 2; ++mi)
#pragma unroll
    for (int j = 0; j < 4; ++j) acc[mi][j] = (v8f){};

  for (int k0 = 0; k0 < K; k0 += 32) {
    if (k0 + 32 < K) {   // warm next K-step of A (global_prefetch_b8)
      __builtin_prefetch(A + (size_t)(m0 + (lane & 15)) * ldA + k0 + 32, 0, 0);
      __builtin_prefetch(A + (size_t)(m0 + 16 + (lane & 15)) * ldA + k0 + 32, 0, 0);
    }
    v16h a0 = frag_a(A, m0, k0, ldA, lane);
    v16h a1 = frag_a(A, m0 + 16, k0, ldA, lane);
#pragma unroll
    for (int j = 0; j < 4; ++j) {
      v16h wf = frag_w(W, n0 + j * 16, k0, ldW, lane);
      acc[0][j] = wmma_f16(a0, wf, acc[0][j]);
      acc[1][j] = wmma_f16(a1, wf, acc[1][j]);
    }
  }
  int nl = lane & 15, mo = (lane & 16) ? 8 : 0;
#pragma unroll
  for (int mi = 0; mi < 2; ++mi)
#pragma unroll
    for (int j = 0; j < 4; ++j)
#pragma unroll
      for (int r = 0; r < 8; ++r)
        Cmat[(size_t)(m0 + mi * 16 + r + mo) * ldC + n0 + j * 16 + nl] = acc[mi][j][r];
}

// dt = softplus(raw + bias)
__global__ void __launch_bounds__(256) dt_kernel(const float* __restrict__ zx,
                                                 const float* __restrict__ dtb,
                                                 float* __restrict__ dt_sp) {
  int idx = blockIdx.x * 256 + threadIdx.x;   // NPTS*NHC
  int n = idx >> 4, h = idx & 15;
  float x = zx[(size_t)n * DPROJ + 2048 + h] + dtb[h];
  dt_sp[idx] = (x > 20.f) ? x : log1pf(__expf(x));
}

// depthwise causal conv(K=4) + SiLU, routed into xh / B / C
__global__ void __launch_bounds__(256) conv_kernel(const float* __restrict__ zx,
                                                   const float* __restrict__ cw,
                                                   const float* __restrict__ cb,
                                                   float* __restrict__ xh32,
                                                   _Float16* __restrict__ xh16,
                                                   _Float16* __restrict__ B16,
                                                   _Float16* __restrict__ C16) {
  int idx = blockIdx.x * 256 + threadIdx.x;   // NPTS*CONVC
  int n = idx / CONVC, cch = idx % CONVC;
  float acc = cb[cch];
#pragma unroll
  for (int k = 0; k < 4; ++k) {
    int m = n - 3 + k;
    if (m >= 0) acc += zx[(size_t)m * DPROJ + DIC + cch] * cw[cch * 4 + k];
  }
  float s = siluf(acc);
  if (cch < DIC) {
    xh32[(size_t)n * DIC + cch] = s;
    xh16[(size_t)n * DIC + cch] = (_Float16)s;
  } else if (cch < 2 * DIC) {
    B16[(size_t)n * DSC + (cch - DIC)] = (_Float16)s;
  } else {
    C16[(size_t)n * DSC + (cch - 2 * DIC)] = (_Float16)s;
  }
}

// per (chunk,head) inclusive cumsum of dt*A (chunk-local log decays)
__global__ void __launch_bounds__(256) cumsum_kernel(const float* __restrict__ dt_sp,
                                                     const float* __restrict__ A_log,
                                                     float* __restrict__ scum) {
  int gid = blockIdx.x * 256 + threadIdx.x;   // NCHUNK*NHC = 2048
  if (gid >= NCHUNK * NHC) return;
  int c = gid >> 4, h = gid & 15;
  float Ah = -__expf(A_log[h]);
  float s = 0.f;
  for (int t = 0; t < QC; ++t) {
    int n = c * QC + t;
    s += dt_sp[n * NHC + h] * Ah;
    scum[n * NHC + h] = s;
  }
}

// ---------------- chunked scan: batched WMMA pieces -------------------------

// G_c = C_chunk @ B_chunk^T (32x32, K=512), parallel over chunks
__global__ void __launch_bounds__(128) gmat_kernel(const _Float16* __restrict__ C16,
                                                   const _Float16* __restrict__ B16,
                                                   float* __restrict__ G) {
  int c = blockIdx.x;
  int wv = threadIdx.x >> 5, lane = threadIdx.x & 31;
  int mt = wv & 1, nt = wv >> 1;
  wave_tile_gemm(C16 + (size_t)c * QC * DSC, B16 + (size_t)c * QC * DSC,
                 G + (size_t)c * QC * QC, mt * 16, nt * 16, DSC, DSC, DSC, QC, lane);
}

// Y_intra = (mask(G) decay dt) @ X, parallel over (chunk, head); stores ys
__global__ void __launch_bounds__(128) intra_kernel(const float* __restrict__ G,
                                                    const float* __restrict__ scum,
                                                    const float* __restrict__ dt_sp,
                                                    const _Float16* __restrict__ xh16,
                                                    float* __restrict__ ys) {
  __shared__ __align__(32) _Float16 Gm[QC][QC];
  __shared__ __align__(32) _Float16 XT[HDC][QC];
  int c = blockIdx.x >> 4, h = blockIdx.x & 15;
  int tid = threadIdx.x, n0 = c * QC;
  for (int i = tid; i < QC * QC; i += 128) {
    int t = i >> 5, tau = i & 31;
    float g = 0.f;
    if (tau <= t) {
      float st = scum[(n0 + t) * NHC + h];
      float sv = scum[(n0 + tau) * NHC + h];
      g = G[(size_t)c * QC * QC + i] * __expf(st - sv) * dt_sp[(n0 + tau) * NHC + h];
    }
    Gm[t][tau] = (_Float16)g;
  }
  for (int i = tid; i < HDC * QC; i += 128) {
    int pp = i >> 5, tau = i & 31;
    XT[pp][tau] = xh16[(size_t)(n0 + tau) * DIC + h * HDC + pp];
  }
  __syncthreads();
  int wv = tid >> 5, lane = tid & 31;
  int mt = wv & 1, pt = wv >> 1;
  v8f acc = {};
  v16h af = frag_a(&Gm[0][0], mt * 16, 0, QC, lane);
  v16h bf = frag_w(&XT[0][0], pt * 16, 0, QC, lane);
  acc = wmma_f16(af, bf, acc);
  int pcol = pt * 16 + (lane & 15);
  int mo = (lane & 16) ? 8 : 0;
#pragma unroll
  for (int r = 0; r < 8; ++r)
    ys[(size_t)(n0 + mt * 16 + r + mo) * DIC + h * HDC + pcol] = acc[r];
}

// Inter-chunk recurrence. 128 blocks = 16 heads x 8 state splits (64 rows each).
// S slice lives in LDS; per chunk: Y_inter += exp(s_t) C.S (atomicAdd),
// then S = exp(s_Q) S + (w (.) B)^T @ X.  B tile is DMA'd to LDS by the TDM.
__global__ void __launch_bounds__(128) seq_scan_kernel(const _Float16* __restrict__ C16,
                                                       const _Float16* __restrict__ B16,
                                                       const _Float16* __restrict__ xh16,
                                                       const float* __restrict__ scum,
                                                       const float* __restrict__ dt_sp,
                                                       float* __restrict__ ys) {
  __shared__ __align__(32) float    S[SLEN][HDC];     // state [s][p]
  __shared__ __align__(32) _Float16 S16T[HDC][SLEN];  // f16 mirror, [p][s]
  __shared__ __align__(32) _Float16 BwT[SLEN][QC];    // (w.B)^T  [s][tau]
  __shared__ __align__(32) _Float16 XT[HDC][QC];      // X^T      [p][tau]
  __shared__ __align__(32) _Float16 Btile[QC][SLEN];  // TDM-staged B tile [tau][s]
  __shared__ float est[QC];
  __shared__ float wta[QC];
  __shared__ float lamS;

  int h  = blockIdx.x / SSPLIT;
  int s0 = (blockIdx.x % SSPLIT) * SLEN;
  int tid = threadIdx.x, lane = tid & 31, wv = tid >> 5;

  for (int i = tid; i < SLEN * HDC; i += 128) {
    S[i >> 5][i & 31] = 0.f;
    S16T[i & 31][i >> 5] = (_Float16)0.f;
  }
  __syncthreads();

  for (int c = 0; c < NCHUNK; ++c) {
    int n0 = c * QC;

#if USE_TDM
    // Tensor Data Mover: load 32x64 f16 tile  B16[n0..n0+31][s0..s0+63]  into Btile.
    if (wv == 0) {
      unsigned long long ga =
          (unsigned long long)(uintptr_t)(B16 + (size_t)n0 * DSC + s0);
      unsigned int ldsoff = (unsigned int)(uintptr_t)&Btile[0][0];  // addr[31:0] = LDS byte offset
      v4u g0;
      g0[0] = 1u;                                   // count=1, user descriptor
      g0[1] = ldsoff;                               // lds_addr
      g0[2] = (unsigned int)ga;                     // global_addr[31:0]
      g0[3] = (unsigned int)((ga >> 32) & 0x1FFFFFFu) | (2u << 30);  // addr[56:32] | type=2
      v8u g1;
      g1[0] = 1u << 16;                             // data_size = 1 (2 bytes)
      g1[1] = (unsigned int)(DSC & 0xFFFF) << 16;   // tensor_dim0[15:0]=512 @ bits[63:48]
      g1[2] = (unsigned int)(NPTS & 0xFFFF) << 16;  // dim0 hi=0 | tensor_dim1[15:0]=4096
      g1[3] = (unsigned int)SLEN << 16;             // dim1 hi=0 | tile_dim0=64
      g1[4] = (unsigned int)QC;                     // tile_dim1=32, tile_dim2=0
      g1[5] = (unsigned int)DSC;                    // tensor_dim0_stride=512 (lo 32)
      g1[6] = 0u;                                   // stride hi | dim1_stride lo
      g1[7] = 0u;
      asm volatile("tensor_load_to_lds %0, %1" :: "s"(g0), "s"(g1) : "memory");
      __builtin_amdgcn_s_wait_tensorcnt(0);
    }
#endif

    if (tid < QC) {
      float sQ = scum[(n0 + QC - 1) * NHC + h];
      float st = scum[(n0 + tid) * NHC + h];
      est[tid] = __expf(st);
      wta[tid] = __expf(sQ - st) * dt_sp[(n0 + tid) * NHC + h];
      if (tid == 0) lamS = __expf(sQ);
    }
    __syncthreads();

    for (int i = tid; i < SLEN * QC; i += 128) {
      int s = i >> 5, tau = i & 31;
#if USE_TDM
      BwT[s][tau] = (_Float16)((float)Btile[tau][s] * wta[tau]);
#else
      BwT[s][tau] = (_Float16)((float)B16[(size_t)(n0 + tau) * DSC + s0 + s] * wta[tau]);
#endif
    }
    for (int i = tid; i < HDC * QC; i += 128) {
      int pp = i >> 5, tau = i & 31;
      XT[pp][tau] = xh16[(size_t)(n0 + tau) * DIC + h * HDC + pp];
    }
    __syncthreads();

    // phase 1: Y_inter tiles, wave wv -> (mt = wv&1, pt = wv>>1), K = SLEN
    {
      int mt = wv & 1, pt = wv >> 1;
      v8f acc = {};
      for (int k0 = 0; k0 < SLEN; k0 += 32) {
        v16h af = frag_a(C16, n0 + mt * 16, s0 + k0, DSC, lane);    // C rows
        v16h bf = frag_w(&S16T[0][0], pt * 16, k0, SLEN, lane);     // S16T[p][s]
        acc = wmma_f16(af, bf, acc);
      }
      int pcol = pt * 16 + (lane & 15);
      int mo = (lane & 16) ? 8 : 0;
#pragma unroll
      for (int r = 0; r < 8; ++r) {
        int t = mt * 16 + r + mo;
        atomicAdd(&ys[(size_t)(n0 + t) * DIC + h * HDC + pcol], acc[r] * est[t]);
      }
    }
    __syncthreads();

    // phase 2: state update, wave wv owns s-tile wv, loops p-tiles. K = QC exact.
    for (int pt = 0; pt < 2; ++pt) {
      int pcol = pt * 16 + (lane & 15);
      int mo = (lane & 16) ? 8 : 0;
      v8f acc;
#pragma unroll
      for (int r = 0; r < 8; ++r) acc[r] = lamS * S[wv * 16 + r + mo][pcol];
      v16h af = frag_a(&BwT[0][0], wv * 16, 0, QC, lane);
      v16h bf = frag_w(&XT[0][0], pt * 16, 0, QC, lane);
      acc = wmma_f16(af, bf, acc);
#pragma unroll
      for (int r = 0; r < 8; ++r) {
        int srow = wv * 16 + r + mo;
        S[srow][pcol] = acc[r];
        S16T[pcol][srow] = (_Float16)acc[r];
      }
    }
    __syncthreads();
  }
}

// y = ys + D*xh ; gated rmsnorm with silu(z) -> f16
__global__ void __launch_bounds__(256) gate_kernel(const float* __restrict__ ys,
                                                   const float* __restrict__ xh32,
                                                   const float* __restrict__ zx,
                                                   const float* __restrict__ Dp,
                                                   const float* __restrict__ gw,
                                                   _Float16* __restrict__ y16) {
  __shared__ float red[256];
  int n = blockIdx.x, tid = threadIdx.x;
  float g0, g1;
  {
    int d = tid;
    float y = ys[(size_t)n * DIC + d] + Dp[d >> 5] * xh32[(size_t)n * DIC + d];
    g0 = y * siluf(zx[(size_t)n * DPROJ + d]);
  }
  {
    int d = tid + 256;
    float y = ys[(size_t)n * DIC + d] + Dp[d >> 5] * xh32[(size_t)n * DIC + d];
    g1 = y * siluf(zx[(size_t)n * DPROJ + d]);
  }
  red[tid] = g0 * g0 + g1 * g1;
  __syncthreads();
  for (int s = 128; s > 0; s >>= 1) { if (tid < s) red[tid] += red[tid + s]; __syncthreads(); }
  float inv = rsqrtf(red[0] / (float)DIC + EPSF);
  y16[(size_t)n * DIC + tid]       = (_Float16)(g0 * inv * gw[tid]);
  y16[(size_t)n * DIC + tid + 256] = (_Float16)(g1 * inv * gw[tid + 256]);
}

// final: resid += hid; rmsnorm; inverse-permute; sigmoid blend with original f
__global__ void __launch_bounds__(256) final_kernel(const float* __restrict__ resid,
                                                    const float* __restrict__ hid,
                                                    const float* __restrict__ fw,
                                                    const float* __restrict__ f,
                                                    const int* __restrict__ rank,
                                                    const float* __restrict__ alpha,
                                                    float* __restrict__ out) {
  __shared__ float red[256];
  int i = blockIdx.x, d = threadIdx.x;
  int r = rank[i];
  float v = resid[(size_t)r * DMC + d] + hid[(size_t)r * DMC + d];
  red[d] = v * v;
  __syncthreads();
  for (int s = 128; s > 0; s >>= 1) { if (d < s) red[d] += red[d + s]; __syncthreads(); }
  float inv = rsqrtf(red[0] / (float)DMC + EPSF);
  float a = 1.f / (1.f + __expf(-alpha[0]));
  out[(size_t)i * DMC + d] = v * inv * fw[d] * a + f[(size_t)i * DMC + d] * (1.f - a);
}

// ---------------- host orchestration ---------------------------------------

static inline char* carve(char*& p, size_t bytes) {
  char* r = p;
  p += (bytes + 255) & ~(size_t)255;
  return r;
}

extern "C" void kernel_launch(void* const* d_in, const int* in_sizes, int n_in,
                              void* d_out, int out_size, void* d_ws, size_t ws_size,
                              hipStream_t stream) {
  const float* p_gs        = (const float*)d_in[1];
  const float* f           = (const float*)d_in[2];
  const float* in_proj_w   = (const float*)d_in[4];   // [L,2064,256]
  const float* conv_w      = (const float*)d_in[5];   // [L,1536,4]
  const float* conv_b      = (const float*)d_in[6];   // [L,1536]
  const float* dt_bias     = (const float*)d_in[7];   // [L,16]
  const float* A_log       = (const float*)d_in[8];   // [L,16]
  const float* Dparam      = (const float*)d_in[9];   // [L,16]
  const float* gnorm_w     = (const float*)d_in[10];  // [L,512]
  const float* out_proj_w  = (const float*)d_in[11];  // [L,256,512]
  const float* blknorm_w   = (const float*)d_in[12];  // [L,256]
  const float* final_w     = (const float*)d_in[13];  // [256]
  const float* alpha       = (const float*)d_in[14];  // [1]
  float* out = (float*)d_out;

  char* p = (char*)d_ws;                              // ~78 MB total
  int*      rank   = (int*)     carve(p, NPTS * 4);
  float*    resid  = (float*)   carve(p, (size_t)NPTS * DMC * 4);
  float*    hid    = (float*)   carve(p, (size_t)NPTS * DMC * 4);
  _Float16* xn16   = (_Float16*)carve(p, (size_t)NPTS * DMC * 2);
  _Float16* Wi16   = (_Float16*)carve(p, (size_t)DPROJ * DMC * 2);
  _Float16* Wo16   = (_Float16*)carve(p, (size_t)DMC * DIC * 2);
  float*    zx     = (float*)   carve(p, (size_t)NPTS * DPROJ * 4);
  float*    xh32   = (float*)   carve(p, (size_t)NPTS * DIC * 4);
  _Float16* xh16   = (_Float16*)carve(p, (size_t)NPTS * DIC * 2);
  _Float16* B16    = (_Float16*)carve(p, (size_t)NPTS * DSC * 2);
  _Float16* C16    = (_Float16*)carve(p, (size_t)NPTS * DSC * 2);
  float*    dt_sp  = (float*)   carve(p, (size_t)NPTS * NHC * 4);
  float*    scum   = (float*)   carve(p, (size_t)NPTS * NHC * 4);
  float*    G      = (float*)   carve(p, (size_t)NCHUNK * QC * QC * 4);
  float*    ysb    = (float*)   carve(p, (size_t)NPTS * DIC * 4);
  _Float16* y16    = (_Float16*)carve(p, (size_t)NPTS * DIC * 2);

  // sort order + permute f into hid (sorted domain)
  rank_kernel<<<NPTS / 256, 256, 0, stream>>>(p_gs, rank);
  scatter_kernel<<<NPTS, 256, 0, stream>>>(f, rank, hid);

  for (int l = 0; l < 2; ++l) {
    // residual + rmsnorm -> xn16
    prenorm_kernel<<<NPTS, 256, 0, stream>>>(hid, resid, blknorm_w + l * DMC, xn16, l > 0);

    // in-projection: [4096,2064] = xn @ Wi^T (K=256): 2x4-blocked main + 1-tile tail
    cvt_kernel<<<(DPROJ * DMC + 255) / 256, 256, 0, stream>>>(
        in_proj_w + (size_t)l * DPROJ * DMC, Wi16, DPROJ * DMC);
    {
      int jobs = (NPTS / 32) * (2048 / 64);            // 128 * 32
      gemm_blocked_kernel<<<(jobs + 7) / 8, 256, 0, stream>>>(
          xn16, Wi16, zx, NPTS / 32, 2048 / 64, DMC, DMC, DMC, DPROJ);
      gemm_kernel<<<(NPTS / 16 + 7) / 8, 256, 0, stream>>>(       // last 16 cols
          xn16, Wi16, zx, NPTS / 16, 1, DMC, DMC, DMC, DPROJ, 2048);
    }

    // dt softplus + depthwise conv + SiLU routing
    dt_kernel<<<NPTS * NHC / 256, 256, 0, stream>>>(zx, dt_bias + l * NHC, dt_sp);
    conv_kernel<<<NPTS * CONVC / 256, 256, 0, stream>>>(
        zx, conv_w + (size_t)l * CONVC * 4, conv_b + (size_t)l * CONVC,
        xh32, xh16, B16, C16);

    // chunked scan
    cumsum_kernel<<<(NCHUNK * NHC + 255) / 256, 256, 0, stream>>>(dt_sp, A_log + l * NHC, scum);
    gmat_kernel<<<NCHUNK, 128, 0, stream>>>(C16, B16, G);
    intra_kernel<<<NCHUNK * NHC, 128, 0, stream>>>(G, scum, dt_sp, xh16, ysb);
    seq_scan_kernel<<<NHC * SSPLIT, 128, 0, stream>>>(C16, B16, xh16, scum, dt_sp, ysb);

    // gated norm -> y16
    gate_kernel<<<NPTS, 256, 0, stream>>>(ysb, xh32, zx, Dparam + l * NHC,
                                          gnorm_w + l * DIC, y16);

    // out-projection: [4096,256] = y @ Wo^T (K=512), 2x4-blocked
    cvt_kernel<<<(DMC * DIC + 255) / 256, 256, 0, stream>>>(
        out_proj_w + (size_t)l * DMC * DIC, Wo16, DMC * DIC);
    {
      int jobs = (NPTS / 32) * (DMC / 64);             // 128 * 4
      gemm_blocked_kernel<<<(jobs + 7) / 8, 256, 0, stream>>>(
          y16, Wo16, hid, NPTS / 32, DMC / 64, DIC, DIC, DIC, DMC);
    }
  }

  // final residual + norm + inverse permute + blend
  final_kernel<<<NPTS, 256, 0, stream>>>(resid, hid, final_w, f, rank, alpha, out);
}
